// ModelBase_25185688224608
// MI455X (gfx1250) — matrix operand order repaired and checked by reading, not compiled, not run on hardware
//
#include <hip/hip_runtime.h>
#include <hip/hip_bf16.h>

typedef __attribute__((ext_vector_type(16))) _Float16 v16h;
typedef __attribute__((ext_vector_type(8)))  _Float16 v8h;
typedef __attribute__((ext_vector_type(8)))  float    v8f;

#define TOKENS      (512 * 1024)
#define TOK_PER_BLK 64
#define KDIM        149
#define KPAD        160        // 5 WMMA K-steps of 32
#define EMB_STRIDE  168        // halves per LDS embed row (160 used + pad)
#define HD          64

// ---------------------------------------------------------------------------
// Pre-pack comb_w[64][149] into per-lane WMMA B-fragment layout, f16, K padded
// to 160 with zeros.  Flat index p = ((kk*4 + n)*32 + lane)*16 + j, where
// b.half[j] = comb_w[o = n*16 + (lane&15)][k = kk*32 + (lane>>4)*8 + (j&7) + (j>>3)*16].
// Total 5*4*32*16 = 10240 halves = 20 KB in d_ws.
// ---------------------------------------------------------------------------
__global__ __launch_bounds__(256) void pack_weights_kernel(
    const float* __restrict__ comb_w, _Float16* __restrict__ wpack) {
  int p = blockIdx.x * 256 + threadIdx.x;
  if (p >= 5 * 4 * 32 * 16) return;
  int j    =  p        & 15;
  int lane = (p >> 4)  & 31;
  int n    = (p >> 9)  & 3;
  int kk   =  p >> 11;
  int o = n * 16 + (lane & 15);
  int k = kk * 32 + (lane >> 4) * 8 + (j & 7) + (j >> 3) * 16;
  float v = (k < KDIM) ? comb_w[o * KDIM + k] : 0.0f;
  wpack[p] = (_Float16)v;
}

__device__ __forceinline__ float fast_sigmoid(float x) {
  return __builtin_amdgcn_rcpf(1.0f + __expf(-x));
}

// ---------------------------------------------------------------------------
// Fused gather + sigmoid-feature + [16x160]x[160x64] WMMA projection.
// 256 threads = 8 wave32; 64 tokens per block.
// ---------------------------------------------------------------------------
__global__ __launch_bounds__(256) void fused_embed_proj_kernel(
    const int* __restrict__ g_test, const int* __restrict__ g_question,
    const int* __restrict__ g_tag,  const int* __restrict__ g_interaction,
    const int* __restrict__ g_testTag,
    const float* __restrict__ fe0,  const float* __restrict__ fe1,
    const float* __restrict__ fe2,  const float* __restrict__ fe3,
    const float* __restrict__ fe4,  const float* __restrict__ fe5,
    const float* __restrict__ fe6,  const float* __restrict__ fe7,
    const float* __restrict__ fe8,  const float* __restrict__ fe9,
    const float* __restrict__ fe10, const float* __restrict__ fe11,
    const float* __restrict__ emb_interaction, const float* __restrict__ emb_test,
    const float* __restrict__ emb_question,    const float* __restrict__ emb_tag,
    const float* __restrict__ emb_testTag,
    const float* __restrict__ lin_w, const float* __restrict__ lin_b,
    const float* __restrict__ comb_b,
    const _Float16* __restrict__ wpack,
    float* __restrict__ out) {
  __shared__ __align__(16) _Float16 s_embed[TOK_PER_BLK * EMB_STRIDE]; // 21504 B
  __shared__ float s_feat[TOK_PER_BLK * 12];                           //  3072 B
  __shared__ int   s_idx [TOK_PER_BLK * 8];                            //  2048 B

  const int tid  = threadIdx.x;
  const int base = blockIdx.x * TOK_PER_BLK;

  // ---- Phase 0: stage per-token indices and scalar features -------------
  if (tid < TOK_PER_BLK) {
    const int tok = base + tid;
    s_idx[tid * 8 + 0] = g_interaction[tok];
    s_idx[tid * 8 + 1] = g_test[tok];
    s_idx[tid * 8 + 2] = g_question[tok];
    s_idx[tid * 8 + 3] = g_tag[tok];
    s_idx[tid * 8 + 4] = g_testTag[tok];
    s_feat[tid * 12 +  0] = fe0[tok];
    s_feat[tid * 12 +  1] = fe1[tok];
    s_feat[tid * 12 +  2] = fe2[tok];
    s_feat[tid * 12 +  3] = fe3[tok];
    s_feat[tid * 12 +  4] = fe4[tok];
    s_feat[tid * 12 +  5] = fe5[tok];
    s_feat[tid * 12 +  6] = fe6[tok];
    s_feat[tid * 12 +  7] = fe7[tok];
    s_feat[tid * 12 +  8] = fe8[tok];
    s_feat[tid * 12 +  9] = fe9[tok];
    s_feat[tid * 12 + 10] = fe10[tok];
    s_feat[tid * 12 + 11] = fe11[tok];
  }
  __syncthreads();

  // ---- Phase A: build embed_f16[64][160] in LDS -------------------------
  // embed layout: [0,84)=ei|et|eq|eg gathers, [84,99)=sigmoid(feat0..2 x5),
  // [99,104)=ett gather, [104,149)=sigmoid(feat3..11 x5), [149,160)=zero pad.
  for (int e = tid; e < TOK_PER_BLK * KPAD; e += 256) {
    const int t = e / KPAD;
    const int k = e - t * KPAD;
    float val;
    if (k < 84) {
      const int sel = k / 21;
      const int col = k - sel * 21;
      const float* tab = (sel == 0) ? emb_interaction
                       : (sel == 1) ? emb_test
                       : (sel == 2) ? emb_question
                                    : emb_tag;
      val = tab[s_idx[t * 8 + sel] * 21 + col];
    } else if (k < 99) {
      const int r = k - 84;
      const int f = r / 5;
      const int jj = r - f * 5;
      val = fast_sigmoid(s_feat[t * 12 + f] * lin_w[jj] + lin_b[jj]);
    } else if (k < 104) {
      val = emb_testTag[s_idx[t * 8 + 4] * 5 + (k - 99)];
    } else if (k < KDIM) {
      const int r = k - 104;
      const int f = r / 5;
      const int jj = r - f * 5;
      val = fast_sigmoid(s_feat[t * 12 + 3 + f] * lin_w[jj] + lin_b[jj]);
    } else {
      val = 0.0f;
    }
    s_embed[t * EMB_STRIDE + k] = (_Float16)val;
  }
  __syncthreads();

  // ---- Phase B: WMMA projection -----------------------------------------
  // wave -> (16-token tile, pair of N=16 output tiles)
  const int wave = tid >> 5;
  const int lane = tid & 31;
  const int tile = wave >> 1;          // 0..3
  const int n0   = (wave & 1) * 2;     // this wave handles n0, n0+1
  const int m    = lane & 15;
  const int h    = lane >> 4;

  v8f acc0 = {};
  v8f acc1 = {};
  // 16-bit A layout (ISA 7.12.2): lane m, half j   -> A[m][h*8 + j]
  //                               lane m, half 8+j -> A[m][16 + h*8 + j]
  const _Float16* arow_base = s_embed + (tile * 16 + m) * EMB_STRIDE + h * 8;
  const v16h* wp = (const v16h*)wpack;

#pragma unroll
  for (int kk = 0; kk < 5; ++kk) {
    const _Float16* arow = arow_base + kk * 32;
    v8h alo = *(const v8h*)(arow);        // ds_load_b128 (16B aligned)
    v8h ahi = *(const v8h*)(arow + 16);   // ds_load_b128
    v16h a = __builtin_shufflevector(alo, ahi,
                                     0, 1, 2, 3, 4, 5, 6, 7,
                                     8, 9, 10, 11, 12, 13, 14, 15);
    v16h b0 = wp[(kk * 4 + n0) * 32 + lane];       // coalesced 32B/lane, WGP$-hot
    v16h b1 = wp[(kk * 4 + n0 + 1) * 32 + lane];
    acc0 = __builtin_amdgcn_wmma_f32_16x16x32_f16(
        false, a, false, b0, (short)0, acc0, false, false);
    acc1 = __builtin_amdgcn_wmma_f32_16x16x32_f16(
        false, a, false, b1, (short)0, acc1, false, false);
  }

  // ---- bias + store (C/D layout: VGPR v, lanes<16 -> M=v, lanes>=16 -> M=v+8)
  const int col0  = n0 * 16 + m;
  const float b0s = comb_b[col0];
  const float b1s = comb_b[col0 + 16];
  const int tok0  = base + tile * 16 + h * 8;
#pragma unroll
  for (int v = 0; v < 8; ++v) {
    float* orow = out + (size_t)(tok0 + v) * HD;
    orow[col0]      = acc0[v] + b0s;
    orow[col0 + 16] = acc1[v] + b1s;
  }
}

// ---------------------------------------------------------------------------
extern "C" void kernel_launch(void* const* d_in, const int* in_sizes, int n_in,
                              void* d_out, int out_size, void* d_ws, size_t ws_size,
                              hipStream_t stream) {
  const int* g_test        = (const int*)d_in[0];
  const int* g_question    = (const int*)d_in[1];
  const int* g_tag         = (const int*)d_in[2];
  // d_in[3] = correct, d_in[4] = mask : unused by the reference output
  const int* g_interaction = (const int*)d_in[5];
  const int* g_testTag     = (const int*)d_in[6];
  const float* fe[12];
  for (int i = 0; i < 12; ++i) fe[i] = (const float*)d_in[7 + i];
  const float* emb_interaction = (const float*)d_in[19];
  const float* emb_test        = (const float*)d_in[20];
  const float* emb_question    = (const float*)d_in[21];
  const float* emb_tag         = (const float*)d_in[22];
  const float* emb_testTag     = (const float*)d_in[23];
  const float* lin_w           = (const float*)d_in[24];
  const float* lin_b           = (const float*)d_in[25];
  const float* comb_w          = (const float*)d_in[26];
  const float* comb_b          = (const float*)d_in[27];

  _Float16* wpack = (_Float16*)d_ws;   // 20 KB of workspace

  pack_weights_kernel<<<40, 256, 0, stream>>>(comb_w, wpack);

  const int blocks = TOKENS / TOK_PER_BLK;   // 8192
  fused_embed_proj_kernel<<<blocks, 256, 0, stream>>>(
      g_test, g_question, g_tag, g_interaction, g_testTag,
      fe[0], fe[1], fe[2], fe[3], fe[4], fe[5], fe[6], fe[7], fe[8], fe[9],
      fe[10], fe[11],
      emb_interaction, emb_test, emb_question, emb_tag, emb_testTag,
      lin_w, lin_b, comb_b, (const _Float16*)wpack, (float*)d_out);
}
